// Block_46806553592205
// MI455X (gfx1250) — compile-verified, matrix operand order checked by reading, not run on hardware
//
#include <hip/hip_runtime.h>
#include <hip/hip_bf16.h>
#include <math.h>

// Problem dims (fixed by reference)
#define WBATCH 4
#define WT     2048
#define WC     512
#define WH     8
#define WD     64
#define WM     (WBATCH * WT)   // 8192 tokens

typedef __bf16 bf16_t;
typedef __attribute__((ext_vector_type(16))) __bf16 v16bf;
typedef __attribute__((ext_vector_type(8)))  __bf16 v8bf;
typedef __attribute__((ext_vector_type(8)))  float  v8f;
typedef __attribute__((ext_vector_type(4)))  int    v4i;
typedef __attribute__((ext_vector_type(8)))  int    v8i;
typedef __attribute__((ext_vector_type(4)))  unsigned int v4u;

// ---- helpers -------------------------------------------------------------

__device__ __forceinline__ bf16_t f2bf(float f) {
  union { float f; unsigned u; } a; a.f = f;
  unsigned r = a.u + 0x7fffu + ((a.u >> 16) & 1u);   // round-nearest-even
  unsigned short s = (unsigned short)(r >> 16);
  bf16_t b; __builtin_memcpy(&b, &s, 2); return b;
}

// Build a 16-bit A/B WMMA fragment from a 32-element row:
// half=0 lanes read elems [0..7] and [16..23]; half=1 lanes read [8..15],[24..31]
__device__ __forceinline__ v16bf mkfrag(const bf16_t* row, int hl) {
  v8bf lo = *(const v8bf*)(row + hl * 8);
  v8bf hi = *(const v8bf*)(row + 16 + hl * 8);
  return __builtin_shufflevector(lo, hi, 0,1,2,3,4,5,6,7,8,9,10,11,12,13,14,15);
}

__device__ __forceinline__ v8f wmma_bf16(v16bf a, v16bf b, v8f c) {
  return __builtin_amdgcn_wmma_f32_16x16x32_bf16(false, a, false, b, (short)0, c,
                                                 false, false);
}

// ---- async global->LDS copy (GLOBAL_LOAD_ASYNC_TO_LDS_B128, ASYNCcnt) ----

#if __has_builtin(__builtin_amdgcn_global_load_async_to_lds_b128)
#define CDNA5_ASYNC_LDS 1
#else
#define CDNA5_ASYNC_LDS 0
#endif

__device__ __forceinline__ void copy16_g2l(const void* g, void* l) {
#if CDNA5_ASYNC_LDS
  __builtin_amdgcn_global_load_async_to_lds_b128(
      (__attribute__((address_space(1))) v4i*)g,
      (__attribute__((address_space(3))) v4i*)l, 0, 0);
#else
  *(uint4*)l = *(const uint4*)g;
#endif
}

__device__ __forceinline__ void async_drain() {
#if CDNA5_ASYNC_LDS
  asm volatile("s_wait_asynccnt 0" ::: "memory");
#endif
}

// ---- Tensor Data Mover: 2D tile load, global -> LDS (TENSORcnt) ----------
// D# layout per CDNA5 ISA ch.8: group0 = {count/flags, lds_addr, global_addr,
// type=2}; group1 = {data_size, tensor dims, tile dims, dim0 stride}.

#if __has_builtin(__builtin_amdgcn_tensor_load_to_lds)
#define CDNA5_TDM 1
#else
#define CDNA5_TDM 0
#endif

#if CDNA5_TDM
__device__ __forceinline__ void tdm_load_2d(unsigned lds_addr, const void* gaddr,
                                            unsigned tensor_d0, unsigned tensor_d1,
                                            unsigned tile_d0, unsigned tile_d1,
                                            unsigned long long stride0_elems) {
  unsigned long long ga = (unsigned long long)gaddr;
  v4u g0;
  g0.x = 1u;                                           // count=1, user mode
  g0.y = lds_addr;                                     // LDS byte address
  g0.z = (unsigned)(ga & 0xffffffffull);               // global_addr[31:0]
  g0.w = (unsigned)((ga >> 32) & 0x1ffffffull) | (2u << 30);  // [56:32] | type=2
  v8i g1;
  g1[0] = 0x00010000;                                  // data_size=1 (2B), mask=0
  g1[1] = (int)((tensor_d0 & 0xffffu) << 16);          // tensor_dim0[15:0]
  g1[2] = (int)((tensor_d0 >> 16) | ((tensor_d1 & 0xffffu) << 16));
  g1[3] = (int)((tensor_d1 >> 16) | (tile_d0 << 16));  // tensor_dim1[31:16]|tile_dim0
  g1[4] = (int)(tile_d1 & 0xffffu);                    // tile_dim1; tile_dim2=0
  g1[5] = (int)(stride0_elems & 0xffffffffull);        // tensor_dim0_stride[31:0]
  g1[6] = (int)((stride0_elems >> 32) & 0xffffull);    // stride[47:32]; dim1_stride=0
  g1[7] = 0;
  v4i z4 = {0, 0, 0, 0};
#if __clang_major__ >= 23
  v8i z8 = {0, 0, 0, 0, 0, 0, 0, 0};
  __builtin_amdgcn_tensor_load_to_lds(g0, g1, z4, z4, z8, 0);
#else
  __builtin_amdgcn_tensor_load_to_lds(g0, g1, z4, z4, 0);
#endif
}
#endif

__device__ __forceinline__ void tensor_drain() {
#if __has_builtin(__builtin_amdgcn_s_wait_tensorcnt)
  __builtin_amdgcn_s_wait_tensorcnt(0);
#else
  asm volatile("s_wait_tensorcnt 0" ::: "memory");
#endif
}

__device__ __forceinline__ unsigned lds_byte_addr(const void* p) {
  return (unsigned)(size_t)(__attribute__((address_space(3))) const char*)p;
}

// ---- elementwise f32 -> bf16 convert ------------------------------------

__global__ void cvt_f32_bf16(const float* __restrict__ in, bf16_t* __restrict__ out, int n) {
  for (int i = blockIdx.x * blockDim.x + threadIdx.x; i < n; i += gridDim.x * blockDim.x)
    out[i] = f2bf(in[i]);
}

// ---- LayerNorm: one wave (32 lanes) per row of 512 -----------------------

__global__ void __launch_bounds__(256) ln_kernel(const float* __restrict__ x,
                                                 const float* __restrict__ g,
                                                 const float* __restrict__ beta,
                                                 bf16_t* __restrict__ out, int rows) {
  int wave = threadIdx.x >> 5, lane = threadIdx.x & 31;
  int row = blockIdx.x * 8 + wave;
  if (row >= rows) return;
  const float* xr = x + (size_t)row * WC;
  float vals[16];
  float s = 0.f;
  #pragma unroll
  for (int k = 0; k < 16; ++k) { vals[k] = xr[lane + 32 * k]; s += vals[k]; }
  #pragma unroll
  for (int o = 16; o >= 1; o >>= 1) s += __shfl_xor(s, o, 32);
  float mean = s * (1.0f / WC);
  float vs = 0.f;
  #pragma unroll
  for (int k = 0; k < 16; ++k) { float d = vals[k] - mean; vs += d * d; }
  #pragma unroll
  for (int o = 16; o >= 1; o >>= 1) vs += __shfl_xor(vs, o, 32);
  float rinv = rsqrtf(vs * (1.0f / WC) + 1e-5f);
  #pragma unroll
  for (int k = 0; k < 16; ++k) {
    int idx = lane + 32 * k;
    out[(size_t)row * WC + idx] = f2bf((vals[k] - mean) * rinv * g[idx] + beta[idx]);
  }
}

// ---- WMMA GEMM: out = epilogue(A[M,K] @ W[K,N] + bias) -------------------
// Workgroup tile 128(M) x 64(N); 8 waves, each wave: 16 rows x four 16x16 tiles.
// A tile is DMA'd by the Tensor Data Mover (wave 0 issues one TENSOR_LOAD_TO_LDS);
// W tile is staged transposed through VGPRs.
// MODE 0: f32 out = acc + bias + resid      (attn-proj / MLP2 with residual)
// MODE 1: bf16 out reshaped to [B,H,T,64]   (Q/K/V)
// MODE 2: bf16 out = gelu(acc + bias)       (MLP1)

template <int MODE>
__global__ void __launch_bounds__(256) gemm_bf16_kernel(
    const bf16_t* __restrict__ A, const bf16_t* __restrict__ W,
    const float* __restrict__ bias, const float* __restrict__ resid,
    float* __restrict__ outF, bf16_t* __restrict__ outB,
    int Mm, int Nn, int Kk) {
  __shared__ bf16_t As[128][32];
  __shared__ bf16_t Bs[64][32];   // transposed: Bs[n][k]
  int tid = threadIdx.x;
  int lane = tid & 31, wave = tid >> 5;
  int hl = lane >> 4, l16 = lane & 15;
  int m0 = blockIdx.y * 128, n0 = blockIdx.x * 64;

  v8f acc[4] = {};
  int nk = Kk / 32;
  for (int kc = 0; kc < nk; ++kc) {
    int k0 = kc * 32;
#if CDNA5_TDM
    if (wave == 0) {   // one TDM descriptor moves the whole 128x32 A tile
      tdm_load_2d(lds_byte_addr(&As[0][0]), A + (size_t)m0 * Kk + k0,
                  (unsigned)Kk, (unsigned)Mm, 32u, 128u,
                  (unsigned long long)Kk);
      tensor_drain();
    }
#else
    {   // fallback: async DMA per lane
      int ar = tid >> 1, ac = (tid & 1) * 16;
      const bf16_t* src = A + (size_t)(m0 + ar) * Kk + k0 + ac;
      copy16_g2l(src,     &As[ar][ac]);
      copy16_g2l(src + 8, &As[ar][ac + 8]);
    }
#endif
    {   // stage W tile transposed: read 8 contiguous n, scatter into Bs[n][k]
      int kk = tid >> 3, ns = (tid & 7) * 8;
      uint4 t = *(const uint4*)(W + (size_t)(k0 + kk) * Nn + n0 + ns);
      const bf16_t* tb = (const bf16_t*)&t;
      #pragma unroll
      for (int i = 0; i < 8; ++i) Bs[ns + i][kk] = tb[i];
    }
    if (kc + 1 < nk) {   // global_prefetch_b8 of next K-chunk
      __builtin_prefetch(A + (size_t)(m0 + (tid >> 1)) * Kk + k0 + 32, 0, 1);
      __builtin_prefetch(W + (size_t)(k0 + 32 + (tid >> 3)) * Nn + n0, 0, 1);
    }
#if !CDNA5_TDM
    async_drain();
#endif
    __syncthreads();
    v16bf af = mkfrag(&As[wave * 16 + l16][0], hl);
    #pragma unroll
    for (int j = 0; j < 4; ++j) {
      v16bf bfv = mkfrag(&Bs[j * 16 + l16][0], hl);
      acc[j] = wmma_bf16(af, bfv, acc[j]);
    }
    __syncthreads();
  }

  // Epilogue. C-tile layout: row = i + 8*(lane>=16), col = lane&15.
  #pragma unroll
  for (int j = 0; j < 4; ++j) {
    #pragma unroll
    for (int i = 0; i < 8; ++i) {
      int row = m0 + wave * 16 + i + hl * 8;
      int col = n0 + j * 16 + l16;
      float v = acc[j][i] + bias[col];
      if constexpr (MODE == 0) {
        v += resid[(size_t)row * Nn + col];
        outF[(size_t)row * Nn + col] = v;
      } else if constexpr (MODE == 1) {
        int b = row / WT, t = row % WT;
        int hh = col >> 6, dd = col & 63;
        outB[((size_t)(b * WH + hh) * WT + t) * WD + dd] = f2bf(v);
      } else {
        float gl = 0.5f * v * (1.0f + erff(v * 0.70710678118f));
        outB[(size_t)row * Nn + col] = f2bf(gl);
      }
    }
  }
}

// ---- Flash-attention: causal, per (b,h), 128 q-rows per workgroup --------

__global__ void __launch_bounds__(256) attn_kernel(
    const bf16_t* __restrict__ Q, const bf16_t* __restrict__ Km,
    const bf16_t* __restrict__ V, bf16_t* __restrict__ Y) {
  __shared__ bf16_t Ks[32][64];       // K rows (key-major)
  __shared__ bf16_t Vs[64][32];       // V transposed: Vs[d][key]
  __shared__ bf16_t Ps[8][16][32];    // per-wave P tile
  int tid = threadIdx.x, lane = tid & 31, wave = tid >> 5;
  int hl = lane >> 4, l16 = lane & 15;
  int bh = blockIdx.y;
  int b = bh >> 3, h = bh & 7;
  int q0 = blockIdx.x * 128;
  const bf16_t* Qp = Q + (size_t)bh * WT * WD;
  const bf16_t* Kp = Km + (size_t)bh * WT * WD;
  const bf16_t* Vp = V + (size_t)bh * WT * WD;

  // Q A-fragments (lane row = l16), d split into two 32-chunks
  int qrow = q0 + wave * 16 + l16;
  v16bf qf[2];
  #pragma unroll
  for (int c = 0; c < 2; ++c) qf[c] = mkfrag(Qp + (size_t)qrow * WD + c * 32, hl);

  v8f oacc[4] = {};
  float mrun[8], srun[8];
  #pragma unroll
  for (int i = 0; i < 8; ++i) { mrun[i] = -3e30f; srun[i] = 0.f; }
  const float scale = 0.125f;   // 1/sqrt(64)
  int nchunk = (q0 + 128) / 32;
  int mymax = q0 + wave * 16 + 15;   // last q row this wave owns

  for (int kc = 0; kc < nchunk; ++kc) {
    int k0 = kc * 32;
    {   // stage K via async DMA to LDS, V transposed via VGPR scatter
      int r = tid >> 3, cs = (tid & 7) * 8;
      copy16_g2l(Kp + (size_t)(k0 + r) * WD + cs, &Ks[r][cs]);
      uint4 t = *(const uint4*)(Vp + (size_t)(k0 + r) * WD + cs);
      const bf16_t* tb = (const bf16_t*)&t;
      #pragma unroll
      for (int i = 0; i < 8; ++i) Vs[cs + i][r] = tb[i];
    }
    async_drain();
    __syncthreads();
    if (k0 <= mymax) {   // wave-uniform predicate (EXEC stays all-ones)
      float st[2][8];
      #pragma unroll
      for (int j = 0; j < 2; ++j) {
        v8f s = {};
        #pragma unroll
        for (int c = 0; c < 2; ++c) {
          v16bf kf = mkfrag(&Ks[j * 16 + l16][c * 32], hl);  // Kᵀ column = K row
          s = wmma_bf16(qf[c], kf, s);
        }
        #pragma unroll
        for (int i = 0; i < 8; ++i) {
          int qr = q0 + wave * 16 + i + hl * 8;
          int kcol = k0 + j * 16 + l16;
          float e = s[i] * scale;
          if (kcol > qr) e = -3e30f;   // causal mask
          st[j][i] = e;
        }
      }
      // online softmax; row lives across the 16 lanes of a half-wave
      #pragma unroll
      for (int i = 0; i < 8; ++i) {
        float mx = fmaxf(st[0][i], st[1][i]);
        #pragma unroll
        for (int o = 8; o >= 1; o >>= 1) mx = fmaxf(mx, __shfl_xor(mx, o, 16));
        float mnew = fmaxf(mrun[i], mx);
        float alpha = __expf(mrun[i] - mnew);
        float p0 = __expf(st[0][i] - mnew);
        float p1 = __expf(st[1][i] - mnew);
        int r = i + hl * 8;
        Ps[wave][r][l16]      = f2bf(p0);
        Ps[wave][r][16 + l16] = f2bf(p1);
        float rs = p0 + p1;
        #pragma unroll
        for (int o = 8; o >= 1; o >>= 1) rs += __shfl_xor(rs, o, 16);
        srun[i] = srun[i] * alpha + rs;
        mrun[i] = mnew;
        #pragma unroll
        for (int n = 0; n < 4; ++n) oacc[n][i] *= alpha;
      }
      // P (C-layout) was spilled to LDS; reload as A-fragment, then P@V
      v16bf pf = mkfrag(&Ps[wave][l16][0], hl);
      #pragma unroll
      for (int n = 0; n < 4; ++n) {
        v16bf vf = mkfrag(&Vs[n * 16 + l16][0], hl);
        oacc[n] = wmma_bf16(pf, vf, oacc[n]);
      }
    }
    __syncthreads();
  }

  #pragma unroll
  for (int n = 0; n < 4; ++n)
    #pragma unroll
    for (int i = 0; i < 8; ++i) {
      int t = q0 + wave * 16 + i + hl * 8;
      float v = oacc[n][i] / srun[i];
      Y[(size_t)(b * WT + t) * WC + h * WD + n * 16 + l16] = f2bf(v);
    }
}

// ---- host launch ---------------------------------------------------------

extern "C" void kernel_launch(void* const* d_in, const int* in_sizes, int n_in,
                              void* d_out, int out_size, void* d_ws, size_t ws_size,
                              hipStream_t stream) {
  (void)in_sizes; (void)n_in; (void)out_size; (void)ws_size;
  const float* x   = (const float*)d_in[0];
  const float* g1  = (const float*)d_in[1];
  const float* be1 = (const float*)d_in[2];
  const float* g2  = (const float*)d_in[3];
  const float* be2 = (const float*)d_in[4];
  const float* Wq = (const float*)d_in[5];  const float* bq = (const float*)d_in[6];
  const float* Wk = (const float*)d_in[7];  const float* bk = (const float*)d_in[8];
  const float* Wv = (const float*)d_in[9];  const float* bv = (const float*)d_in[10];
  const float* Wp = (const float*)d_in[11]; const float* bp = (const float*)d_in[12];
  const float* W1 = (const float*)d_in[13]; const float* b1 = (const float*)d_in[14];
  const float* W2 = (const float*)d_in[15]; const float* b2 = (const float*)d_in[16];
  float* out = (float*)d_out;

  char* ws = (char*)d_ws;
  size_t off = 0;
  auto alloc = [&](size_t bytes) -> char* {
    char* p = ws + off; off += (bytes + 255) & ~(size_t)255; return p;
  };
  const size_t CC = (size_t)WC * WC;          // 262144
  const size_t C4 = (size_t)WC * 4 * WC;      // 1048576
  bf16_t* wqb = (bf16_t*)alloc(CC * 2);
  bf16_t* wkb = (bf16_t*)alloc(CC * 2);
  bf16_t* wvb = (bf16_t*)alloc(CC * 2);
  bf16_t* wpb = (bf16_t*)alloc(CC * 2);
  bf16_t* w1b = (bf16_t*)alloc(C4 * 2);
  bf16_t* w2b = (bf16_t*)alloc(C4 * 2);
  bf16_t* h1  = (bf16_t*)alloc((size_t)WM * WC * 2);
  bf16_t* qb  = (bf16_t*)alloc((size_t)WM * WC * 2);
  bf16_t* kb  = (bf16_t*)alloc((size_t)WM * WC * 2);
  bf16_t* vb  = (bf16_t*)alloc((size_t)WM * WC * 2);
  bf16_t* yb  = (bf16_t*)alloc((size_t)WM * WC * 2);
  float*  x1  = (float*)alloc((size_t)WM * WC * 4);
  bf16_t* h2  = (bf16_t*)alloc((size_t)WM * WC * 2);
  bf16_t* act = (bf16_t*)alloc((size_t)WM * 4 * WC * 2);

  // 1) weights -> bf16
  cvt_f32_bf16<<<1024, 256, 0, stream>>>(Wq, wqb, (int)CC);
  cvt_f32_bf16<<<1024, 256, 0, stream>>>(Wk, wkb, (int)CC);
  cvt_f32_bf16<<<1024, 256, 0, stream>>>(Wv, wvb, (int)CC);
  cvt_f32_bf16<<<1024, 256, 0, stream>>>(Wp, wpb, (int)CC);
  cvt_f32_bf16<<<2048, 256, 0, stream>>>(W1, w1b, (int)C4);
  cvt_f32_bf16<<<2048, 256, 0, stream>>>(W2, w2b, (int)C4);

  // 2) LN1
  ln_kernel<<<WM / 8, 256, 0, stream>>>(x, g1, be1, h1, WM);

  // 3) Q/K/V projections (head-reshaped bf16 outputs)
  dim3 gq(WC / 64, WM / 128);                   // (8, 64)
  gemm_bf16_kernel<1><<<gq, 256, 0, stream>>>(h1, wqb, bq, nullptr, nullptr, qb, WM, WC, WC);
  gemm_bf16_kernel<1><<<gq, 256, 0, stream>>>(h1, wkb, bk, nullptr, nullptr, kb, WM, WC, WC);
  gemm_bf16_kernel<1><<<gq, 256, 0, stream>>>(h1, wvb, bv, nullptr, nullptr, vb, WM, WC, WC);

  // 4) causal flash attention
  dim3 ga(WT / 128, WBATCH * WH);               // (16, 32)
  attn_kernel<<<ga, 256, 0, stream>>>(qb, kb, vb, yb);

  // 5) output projection + residual -> x1 (f32)
  gemm_bf16_kernel<0><<<gq, 256, 0, stream>>>(yb, wpb, bp, x, x1, nullptr, WM, WC, WC);

  // 6) LN2
  ln_kernel<<<WM / 8, 256, 0, stream>>>(x1, g2, be2, h2, WM);

  // 7) MLP up + exact GELU -> bf16
  dim3 gm(4 * WC / 64, WM / 128);               // (32, 64)
  gemm_bf16_kernel<2><<<gm, 256, 0, stream>>>(h2, w1b, b1, nullptr, nullptr, act, WM, 4 * WC, WC);

  // 8) MLP down + residual -> d_out (f32)
  gemm_bf16_kernel<0><<<gq, 256, 0, stream>>>(act, w2b, b2, x1, out, nullptr, WM, WC, 4 * WC);
}
